// SlidingWindowAttention_1460288881457
// MI455X (gfx1250) — matrix operand order repaired and checked by reading, not compiled, not run on hardware
//
#include <hip/hip_runtime.h>
#include <hip/hip_bf16.h>

// CDNA5 / gfx1250 sliding-window attention, bf16 WMMA path.
// B=2, N=4096, C=1024, H=16, D=64, window half-size 128, key window 384.

typedef __attribute__((ext_vector_type(16))) __bf16 v16bf;
typedef __attribute__((ext_vector_type(8)))  __bf16 v8bf;
typedef __attribute__((ext_vector_type(4)))  __bf16 v4bf;
typedef __attribute__((ext_vector_type(8)))  float  v8f;

#define WMMA_BF16(A, B, C) \
  __builtin_amdgcn_wmma_f32_16x16x32_bf16(false, (A), false, (B), (short)0, (C), false, false)

// Build a 16-element bf16 fragment from two contiguous 16-byte chunks.
static __device__ __forceinline__ v16bf ld_frag(const __bf16* p0, const __bf16* p1) {
  union { v16bf v; v8bf h[2]; } u;
  u.h[0] = *(const v8bf*)p0;
  u.h[1] = *(const v8bf*)p1;
  return u.v;
}

// ---------------- conversion kernels ----------------

// 4 elements/thread: b128 read, 8B write.
__global__ void cvt4_kernel(const float* __restrict__ in, __bf16* __restrict__ out, int n4) {
  int i = blockIdx.x * 256 + threadIdx.x;
  if (i < n4) {
    float4 f = ((const float4*)in)[i];
    v4bf o;
    o[0] = (__bf16)f.x; o[1] = (__bf16)f.y; o[2] = (__bf16)f.z; o[3] = (__bf16)f.w;
    *(v4bf*)(out + 4 * (size_t)i) = o;
  }
}

// in: [K][Ncols] f32 -> out: [Ncols][K] bf16 (LDS-tiled 32x32, coalesced both ways).
__global__ __launch_bounds__(256) void tcvt_kernel(const float* __restrict__ in,
                                                   __bf16* __restrict__ out,
                                                   int K, int Ncols) {
  __shared__ float tile[32][33];
  int tx = threadIdx.x;        // 0..31
  int ty = threadIdx.y;        // 0..7
  int n0 = blockIdx.x * 32;    // along Ncols
  int k0 = blockIdx.y * 32;    // along K
#pragma unroll
  for (int i = 0; i < 4; ++i)
    tile[ty + 8 * i][tx] = in[(size_t)(k0 + ty + 8 * i) * Ncols + n0 + tx];
  __syncthreads();
#pragma unroll
  for (int i = 0; i < 4; ++i)
    out[(size_t)(n0 + ty + 8 * i) * K + k0 + tx] = (__bf16)tile[tx][ty + 8 * i];
}

// ---------------- QKV GEMM: [8192x1024] x [1024x3072] ----------------
// 8 waves per block; wave tile 64x64 (4x4 WMMA accs, 16 WMMA : 16 b128 loads
// per K-step); block tile 128x256. Epilogue scatters into q/k/v [B*H][N][D].

__global__ __launch_bounds__(256) void gemm_qkv_kernel(
    const __bf16* __restrict__ A,   // [8192][1024] bf16 (x)
    const __bf16* __restrict__ BT,  // [3072][1024] bf16 (w_qkv^T)
    __bf16* __restrict__ qd, __bf16* __restrict__ kd, __bf16* __restrict__ vd) {
  const int K = 1024;
  int tid   = threadIdx.x;
  int w     = tid >> 5, lane = tid & 31;
  int half  = lane >> 4, colIn = lane & 15;
  int kb    = half * 8;
  int waveM = w & 1, waveN = w >> 1;
  int m0 = blockIdx.x * 128 + waveM * 64;
  int n0 = blockIdx.y * 256 + waveN * 64;

  v8f acc[4][4];
#pragma unroll
  for (int i = 0; i < 4; ++i)
#pragma unroll
    for (int j = 0; j < 4; ++j) acc[i][j] = {};

  const __bf16* arow[4];
  const __bf16* brow[4];
#pragma unroll
  for (int i = 0; i < 4; ++i) arow[i] = A + (size_t)(m0 + 16 * i + colIn) * K;
#pragma unroll
  for (int j = 0; j < 4; ++j) brow[j] = BT + (size_t)(n0 + 16 * j + colIn) * K;

  for (int kk = 0; kk < K; kk += 32) {
    v16bf a[4];
#pragma unroll
    for (int i = 0; i < 4; ++i)
      a[i] = ld_frag(arow[i] + kk + kb, arow[i] + kk + kb + 16);
#pragma unroll
    for (int j = 0; j < 4; ++j) {
      const __bf16* bp = brow[j] + kk + kb;
      v16bf b = ld_frag(bp, bp + 16);
#pragma unroll
      for (int i = 0; i < 4; ++i) acc[i][j] = WMMA_BF16(a[i], b, acc[i][j]);
    }
  }

#pragma unroll
  for (int i = 0; i < 4; ++i)
#pragma unroll
    for (int j = 0; j < 4; ++j)
#pragma unroll
      for (int r = 0; r < 8; ++r) {
        int row = m0 + 16 * i + r + 8 * half;    // 0..8191
        int col = n0 + 16 * j + colIn;           // 0..3071
        int t = col >> 10, c = col & 1023;
        int hh = c >> 6, d = c & 63;
        int bI = row >> 12, nr = row & 4095;
        __bf16 val = (__bf16)acc[i][j][r];
        size_t idx = ((size_t)(bI * 16 + hh) * 4096 + nr) * 64 + d;
        if (t == 0) qd[idx] = val;
        else if (t == 1) kd[idx] = val;
        else vd[idx] = val;
      }
}

// ---------------- fused sliding-window attention ----------------
// grid: (nb=32 blocks, B*H=32). 256 threads = 8 waves; wave w owns q rows
// 16w..16w+15 of the 128-row block. Online (flash) softmax over 12 chunks
// of 32 keys; V^T staged in LDS; P re-laid out through per-wave LDS.

__global__ __launch_bounds__(256) void attn_kernel(
    const __bf16* __restrict__ q, const __bf16* __restrict__ kkb,
    const __bf16* __restrict__ vvb, __bf16* __restrict__ aout) {
  const int N = 4096;
  __shared__ __align__(16) __bf16 Vt[64][392];     // V^T: [d][key], ~50 KB
  __shared__ __align__(16) __bf16 Pst[8][16][40];  // per-wave P staging, 10 KB

  int tid = threadIdx.x;
  int blk = blockIdx.x;   // query block 0..31
  int bh  = blockIdx.y;   // b*16+h
  int h = bh & 15, bIdx = bh >> 4;
  const __bf16* qh = q   + (size_t)bh * N * 64;
  const __bf16* kh = kkb + (size_t)bh * N * 64;
  const __bf16* vh = vvb + (size_t)bh * N * 64;
  int blkBase = blk * 128;

  // stage V^T (zero-fill out-of-range keys)
  for (int e = tid; e < 384 * 64; e += 256) {
    int key = e >> 6, d = e & 63;
    int kp = blkBase - 128 + key;
    __bf16 val = (__bf16)0.0f;
    if (kp >= 0 && kp < N) val = vh[(size_t)kp * 64 + d];
    Vt[d][key] = val;
  }
  __syncthreads();

  int w = tid >> 5, lane = tid & 31;
  int half = lane >> 4, colIn = lane & 15;
  int kb = half * 8;

  // Q A-fragments for d=0..31 and d=32..63 (lane = M row)
  const __bf16* qbase = qh + (size_t)(blkBase + 16 * w + colIn) * 64;
  v16bf aq0 = ld_frag(qbase + kb, qbase + kb + 16);
  v16bf aq1 = ld_frag(qbase + 32 + kb, qbase + 32 + kb + 16);

  float m[8], l[8];
  v8f o[4];
#pragma unroll
  for (int r = 0; r < 8; ++r) { m[r] = -1e30f; l[r] = 0.0f; }
#pragma unroll
  for (int j = 0; j < 4; ++j) o[j] = {};

  const float scale = 0.125f;  // 64^-0.5
  v16bf zf = {};

  for (int t = 0; t < 12; ++t) {
    int kc = 32 * t;
    v8f s0 = {}, s1 = {};
#pragma unroll
    for (int u = 0; u < 2; ++u) {
      int kpL = blkBase - 128 + kc + 16 * u + colIn;  // this lane's key column
      v16bf b0 = zf, b1 = zf;
      if (kpL >= 0 && kpL < N) {
        const __bf16* kp0 = kh + (size_t)kpL * 64 + kb;
        b0 = ld_frag(kp0, kp0 + 16);         // d = kb..kb+7, 16+kb..16+kb+7
        b1 = ld_frag(kp0 + 32, kp0 + 48);    // d = 32+kb..+7, 48+kb..+7
      }
      if (u == 0) { s0 = WMMA_BF16(aq0, b0, s0); s0 = WMMA_BF16(aq1, b1, s0); }
      else        { s1 = WMMA_BF16(aq0, b0, s1); s1 = WMMA_BF16(aq1, b1, s1); }
    }

    // scale + window mask (row = 16w + r + 8*half, col = kc + 16u + colIn)
#pragma unroll
    for (int r = 0; r < 8; ++r) {
      int rowg = 16 * w + r + 8 * half;
      int col0 = kc + colIn, col1 = kc + 16 + colIn;
      int d0 = col0 - 128 - rowg, d1 = col1 - 128 - rowg;
      int kp0 = blkBase + col0 - 128, kp1 = blkBase + col1 - 128;
      bool ok0 = (d0 >= -128) && (d0 <= 128) && (kp0 >= 0) && (kp0 < N);
      bool ok1 = (d1 >= -128) && (d1 <= 128) && (kp1 >= 0) && (kp1 < N);
      s0[r] = ok0 ? s0[r] * scale : -1e30f;
      s1[r] = ok1 ? s1[r] * scale : -1e30f;
    }

    // online softmax: row reductions across the 16 lanes of each half
    float p0[8], p1[8], corr[8];
#pragma unroll
    for (int r = 0; r < 8; ++r) {
      float mx = fmaxf(s0[r], s1[r]);
      mx = fmaxf(mx, __shfl_xor(mx, 1, 16));
      mx = fmaxf(mx, __shfl_xor(mx, 2, 16));
      mx = fmaxf(mx, __shfl_xor(mx, 4, 16));
      mx = fmaxf(mx, __shfl_xor(mx, 8, 16));
      float mn = fmaxf(m[r], mx);
      float c = __expf(m[r] - mn);
      m[r] = mn;
      corr[r] = c;
      p0[r] = __expf(s0[r] - mn);
      p1[r] = __expf(s1[r] - mn);
      float sum = p0[r] + p1[r];
      sum += __shfl_xor(sum, 1, 16);
      sum += __shfl_xor(sum, 2, 16);
      sum += __shfl_xor(sum, 4, 16);
      sum += __shfl_xor(sum, 8, 16);
      l[r] = l[r] * c + sum;
    }
#pragma unroll
    for (int j = 0; j < 4; ++j)
#pragma unroll
      for (int r = 0; r < 8; ++r) o[j][r] *= corr[r];

    // C-layout -> A-layout relayout of P via per-wave LDS (in-order per wave)
#pragma unroll
    for (int r = 0; r < 8; ++r) {
      int row = r + 8 * half;
      Pst[w][row][colIn]      = (__bf16)p0[r];
      Pst[w][row][16 + colIn] = (__bf16)p1[r];
    }
    const __bf16* pr = &Pst[w][colIn][0];
    v16bf pa = ld_frag(pr + kb, pr + kb + 16);

    // O += P(16x32) @ V(32x64): 4 WMMAs, B-frags from transposed V in LDS
#pragma unroll
    for (int j = 0; j < 4; ++j) {
      const __bf16* vb = &Vt[16 * j + colIn][kc + kb];
      v16bf bv = ld_frag(vb, vb + 16);
      o[j] = WMMA_BF16(pa, bv, o[j]);
    }
  }

  // normalize and write [B][N][C] bf16 for the output projection
#pragma unroll
  for (int r = 0; r < 8; ++r) {
    float inv = 1.0f / l[r];
    int n = blkBase + 16 * w + r + 8 * half;
#pragma unroll
    for (int j = 0; j < 4; ++j) {
      int c = h * 64 + 16 * j + colIn;
      aout[(size_t)(bIdx * N + n) * 1024 + c] = (__bf16)(o[j][r] * inv);
    }
  }
}

// ---------------- output projection: [8192x1024] x [1024x1024] + bias ----------------

__global__ __launch_bounds__(256) void gemm_proj_kernel(
    const __bf16* __restrict__ A,   // attn out bf16 [8192][1024]
    const __bf16* __restrict__ BT,  // w_proj^T bf16 [1024][1024]
    const float* __restrict__ bias, float* __restrict__ out) {
  const int K = 1024;
  int tid = threadIdx.x;
  int w = tid >> 5, lane = tid & 31;
  int half = lane >> 4, colIn = lane & 15;
  int kb = half * 8;
  int waveM = w & 1, waveN = w >> 1;
  int m0 = blockIdx.x * 128 + waveM * 64;
  int n0 = blockIdx.y * 256 + waveN * 64;

  v8f acc[4][4];
#pragma unroll
  for (int i = 0; i < 4; ++i)
#pragma unroll
    for (int j = 0; j < 4; ++j) acc[i][j] = {};

  const __bf16* arow[4];
  const __bf16* brow[4];
#pragma unroll
  for (int i = 0; i < 4; ++i) arow[i] = A + (size_t)(m0 + 16 * i + colIn) * K;
#pragma unroll
  for (int j = 0; j < 4; ++j) brow[j] = BT + (size_t)(n0 + 16 * j + colIn) * K;

  for (int kk = 0; kk < K; kk += 32) {
    v16bf a[4];
#pragma unroll
    for (int i = 0; i < 4; ++i)
      a[i] = ld_frag(arow[i] + kk + kb, arow[i] + kk + kb + 16);
#pragma unroll
    for (int j = 0; j < 4; ++j) {
      const __bf16* bp = brow[j] + kk + kb;
      v16bf b = ld_frag(bp, bp + 16);
#pragma unroll
      for (int i = 0; i < 4; ++i) acc[i][j] = WMMA_BF16(a[i], b, acc[i][j]);
    }
  }

#pragma unroll
  for (int i = 0; i < 4; ++i)
#pragma unroll
    for (int j = 0; j < 4; ++j)
#pragma unroll
      for (int r = 0; r < 8; ++r) {
        int row = m0 + 16 * i + r + 8 * half;
        int col = n0 + 16 * j + colIn;
        out[(size_t)row * 1024 + col] = acc[i][j][r] + bias[col];
      }
}

// ---------------- host launch ----------------

extern "C" void kernel_launch(void* const* d_in, const int* in_sizes, int n_in,
                              void* d_out, int out_size, void* d_ws, size_t ws_size,
                              hipStream_t stream) {
  (void)in_sizes; (void)n_in; (void)out_size; (void)ws_size;
  const float* x      = (const float*)d_in[0];  // [2,4096,1024]
  const float* w_qkv  = (const float*)d_in[1];  // [1024,3072]
  const float* w_proj = (const float*)d_in[2];  // [1024,1024]
  const float* b_proj = (const float*)d_in[3];  // [1024]
  float* out = (float*)d_out;

  char* ws = (char*)d_ws;
  const size_t MB = 1u << 20;
  __bf16* xb     = (__bf16*)(ws + 0 * MB);   // 16 MB
  __bf16* wqkvT  = (__bf16*)(ws + 16 * MB);  // 6 MB
  __bf16* wprojT = (__bf16*)(ws + 22 * MB);  // 2 MB
  __bf16* qd     = (__bf16*)(ws + 24 * MB);  // 16 MB
  __bf16* kd     = (__bf16*)(ws + 40 * MB);  // 16 MB
  __bf16* vd     = (__bf16*)(ws + 56 * MB);  // 16 MB
  __bf16* ao     = (__bf16*)(ws + 72 * MB);  // 16 MB  (total 88 MB)

  int n4 = (8192 * 1024) / 4;
  cvt4_kernel<<<(n4 + 255) / 256, 256, 0, stream>>>(x, xb, n4);
  tcvt_kernel<<<dim3(3072 / 32, 1024 / 32), dim3(32, 8), 0, stream>>>(w_qkv, wqkvT, 1024, 3072);
  tcvt_kernel<<<dim3(1024 / 32, 1024 / 32), dim3(32, 8), 0, stream>>>(w_proj, wprojT, 1024, 1024);

  gemm_qkv_kernel<<<dim3(64, 12), 256, 0, stream>>>(xb, wqkvT, qd, kd, vd);
  attn_kernel<<<dim3(32, 32), 256, 0, stream>>>(qd, kd, vd, ao);
  gemm_proj_kernel<<<dim3(64, 4), 256, 0, stream>>>(ao, wprojT, b_proj, out);
}
